// ModifiedDynamicGRU_28776280883491
// MI455X (gfx1250) — matrix operand order repaired and checked by reading, not compiled, not run on hardware
//
#include <hip/hip_runtime.h>
#include <hip/hip_bf16.h>

// ---------------------------------------------------------------------------
// ModifiedDynamicGRU for MI455X (gfx1250, wave32, WMMA bf16 16x16x32)
// B=1024, T=200, D=128, U=128, G=3U=384
//
// One workgroup (256 threads = 8 waves) owns a 16-row batch tile for the
// whole T=200 recurrence. Weights are converted once to bf16 in LDS, then
// all 24 weight WMMA fragments are hoisted into registers (loop-invariant),
// so the steady-state step does only 16 ds_load_b128 (x_t and h tiles),
// 24 WMMAs, and register-resident gate math. The next step's x-tile global
// load is software-pipelined across the current step's GEMM.
// ---------------------------------------------------------------------------

typedef __attribute__((ext_vector_type(16))) __bf16 v16bf;
typedef __attribute__((ext_vector_type(8)))  float  v8f;

namespace {

constexpr int Bc = 1024;
constexpr int Tc = 200;
constexpr int Dc = 128;   // input dim  (== K for x GEMM)
constexpr int Uc = 128;   // hidden dim (== K for h GEMM)
constexpr int Gc = 384;   // 3 * U
constexpr int BTILE = 16; // batch rows per workgroup

// bank-conflict swizzle: rotate K within a 128-element row by 16*(row%8)
__device__ __forceinline__ int swz(int row, int k) {
  return row * 128 + ((k + ((row & 7) << 4)) & 127);
}

__device__ __forceinline__ unsigned short f2bf(float f) {
  unsigned int u = __builtin_bit_cast(unsigned int, f);
  unsigned int r = u + 0x7FFFu + ((u >> 16) & 1u); // round to nearest even
  return (unsigned short)(r >> 16);
}

struct Pack32 { uint4 lo; uint4 hi; };

// load a 32-byte WMMA operand fragment from two 16B LDS chunks
__device__ __forceinline__ v16bf ld_frag(const unsigned short* s, int o0, int o1) {
  Pack32 p;
  p.lo = *reinterpret_cast<const uint4*>(s + o0);
  p.hi = *reinterpret_cast<const uint4*>(s + o1);
  return __builtin_bit_cast(v16bf, p);
}

__device__ __forceinline__ v8f vbc(float v) {
  v8f c;
#pragma unroll
  for (int i = 0; i < 8; ++i) c[i] = v;
  return c;
}

__device__ __forceinline__ float fast_sigmoid(float v) {
  return 1.0f / (1.0f + __expf(-v));
}

__device__ __forceinline__ float fast_tanh(float v) {
  float e = __expf(2.0f * v);          // tanh(v) = 1 - 2/(e^{2v}+1)
  return 1.0f - 2.0f / (e + 1.0f);
}

} // namespace

__global__ __launch_bounds__(256, 1)
void gru_fused_kernel(const float* __restrict__ xin,   // (B, T, D)
                      const int*   __restrict__ seq,   // (B, 1)
                      const float* __restrict__ Wf,    // (D, 3U) row-major
                      const float* __restrict__ Rf,    // (U, 3U) row-major
                      const float* __restrict__ bias,  // (2, 3U)
                      float*       __restrict__ out)   // (B, T, U)
{
  __shared__ unsigned short sW[Gc * Dc];    // 96 KB staging (bf16, col-major+swz)
  __shared__ unsigned short sR[Gc * Uc];    // 96 KB staging
  __shared__ unsigned short sX[BTILE * Dc]; // 4 KB, x_t tile bf16
  __shared__ unsigned short sH[BTILE * Uc]; // 4 KB, h tile bf16

  const int tid  = threadIdx.x;
  const int wave = tid >> 5;
  const int lane = tid & 31;
  const int hi   = (lane >> 4) & 1;
  const int lm   = lane & 15;
  const int col  = wave * 16 + lm;            // hidden-unit column this lane owns
  const int b0   = blockIdx.x * BTILE;

  // --- one-time: convert weights f32 -> bf16, transposed (column-major) ---
  for (int i = tid; i < Dc * Gc; i += 256) {
    int k = i / Gc;          // K index (contiguous-128 dim in LDS)
    int n = i - k * Gc;      // output column 0..383
    int a = swz(n, k);
    sW[a] = f2bf(Wf[i]);
    sR[a] = f2bf(Rf[i]);
  }
  for (int i = tid; i < BTILE * Uc; i += 256) sH[i] = 0;
  __syncthreads();

  // --- hoist all 24 loop-invariant weight fragments into registers ---
  v16bf fW[12], fR[12];                       // [gate*4 + kk]
#pragma unroll
  for (int g = 0; g < 3; ++g) {
#pragma unroll
    for (int kk = 0; kk < 4; ++kk) {
      const int n    = g * 128 + col;
      const int b_lo = kk * 32 + (hi ? 16 : 0);
      const int o    = swz(n, b_lo);
      fW[g * 4 + kk] = ld_frag(sW, o, o + 8);
      fR[g * 4 + kk] = ld_frag(sR, o, o + 8);
    }
  }

  // per-lane constants
  const float binz = bias[col],       binr = bias[128 + col], binh = bias[256 + col];
  const float brcz = bias[384 + col], brcr = bias[512 + col], brch = bias[640 + col];

  int slen[8];
#pragma unroll
  for (int r = 0; r < 8; ++r) slen[r] = seq[b0 + r + 8 * hi];

  float hreg[8], prev[8];
#pragma unroll
  for (int r = 0; r < 8; ++r) { hreg[r] = 0.0f; prev[r] = 0.0f; }

  // --- software pipeline: pre-load x-tile for t = 0 ---
  const int xrow = (tid * 8) >> 7;            // fixed per thread (8 elems/thread)
  const int xk   = (tid * 8) & 127;
  const float* xbase = xin + ((size_t)(b0 + xrow) * Tc) * Dc + xk;
  float4 nf0 = reinterpret_cast<const float4*>(xbase)[0];
  float4 nf1 = reinterpret_cast<const float4*>(xbase)[1];

  for (int t = 0; t < Tc; ++t) {
    // --- stage x_t (already in registers) into LDS as bf16 ---
    {
      union { unsigned short u16[8]; uint4 v; } pk;
      pk.u16[0] = f2bf(nf0.x); pk.u16[1] = f2bf(nf0.y);
      pk.u16[2] = f2bf(nf0.z); pk.u16[3] = f2bf(nf0.w);
      pk.u16[4] = f2bf(nf1.x); pk.u16[5] = f2bf(nf1.y);
      pk.u16[6] = f2bf(nf1.z); pk.u16[7] = f2bf(nf1.w);
      *reinterpret_cast<uint4*>(&sX[swz(xrow, xk)]) = pk.v;
    }
    // --- issue global loads for t+1 now; they complete during this step ---
    {
      const int tn = (t + 1 < Tc) ? t + 1 : t;   // branch-free clamp
      const float4* g = reinterpret_cast<const float4*>(xbase + (size_t)tn * Dc);
      nf0 = g[0];
      nf1 = g[1];
    }
    __syncthreads(); // x_t staged; previous step's sH writes visible

    // --- dual GEMM: acc = bias; acc += x_t@W ; acc += h@R (bf16 WMMA) ---
    v8f xz = vbc(binz), xr = vbc(binr), xh = vbc(binh);
    v8f rz = vbc(brcz), rr = vbc(brcr), rh = vbc(brch);

#pragma unroll
    for (int kk = 0; kk < 4; ++kk) {
      const int k0   = kk * 32;
      const int a_lo = k0 + (hi ? 16 : 0);   // A frag: K halves per half-wave
      v16bf ax = ld_frag(sX, swz(lm, a_lo), swz(lm, a_lo + 8));
      v16bf ah = ld_frag(sH, swz(lm, a_lo), swz(lm, a_lo + 8));

      xz = __builtin_amdgcn_wmma_f32_16x16x32_bf16(false, ax, false, fW[0 * 4 + kk], (short)0, xz, false, false);
      xr = __builtin_amdgcn_wmma_f32_16x16x32_bf16(false, ax, false, fW[1 * 4 + kk], (short)0, xr, false, false);
      xh = __builtin_amdgcn_wmma_f32_16x16x32_bf16(false, ax, false, fW[2 * 4 + kk], (short)0, xh, false, false);
      rz = __builtin_amdgcn_wmma_f32_16x16x32_bf16(false, ah, false, fR[0 * 4 + kk], (short)0, rz, false, false);
      rr = __builtin_amdgcn_wmma_f32_16x16x32_bf16(false, ah, false, fR[1 * 4 + kk], (short)0, rr, false, false);
      rh = __builtin_amdgcn_wmma_f32_16x16x32_bf16(false, ah, false, fR[2 * 4 + kk], (short)0, rh, false, false);
    }
    __syncthreads(); // all waves finished reading sX / sH

    // --- gates + state update, all in registers (z/r/h columns co-resident) ---
#pragma unroll
    for (int r = 0; r < 8; ++r) {
      const int row = r + 8 * hi;             // C/D layout: VGPR r, half-wave
      float z  = fast_sigmoid(xz[r] + rz[r]);
      float rg = fast_sigmoid(xr[r] + rr[r]);
      float hh = fast_tanh(xh[r] + rg * rh[r]);
      float hc = hreg[r];
      float hn = z * hc + (1.0f - z) * hh;
      bool  m  = t < slen[r];
      float hnew = m ? hn : hc;
      float ov   = m ? hn : prev[r];
      hreg[r] = hnew;
      prev[r] = ov;
      out[((size_t)(b0 + row) * Tc + t) * Uc + col] = ov;
      sH[swz(row, col)] = f2bf(hnew);
    }
    // next iteration's post-staging barrier protects these sH writes
  }
}

extern "C" void kernel_launch(void* const* d_in, const int* in_sizes, int n_in,
                              void* d_out, int out_size, void* d_ws, size_t ws_size,
                              hipStream_t stream) {
  const float* rnn_input = (const float*)d_in[0];
  const int*   seq_len   = (const int*)d_in[1];
  const float* kernel    = (const float*)d_in[2];
  const float* rkernel   = (const float*)d_in[3];
  const float* bias      = (const float*)d_in[4];
  float* out = (float*)d_out;

  dim3 grid(Bc / BTILE); // 64 workgroups, one 16-row batch tile each
  dim3 block(256);       // 8 waves (wave32)
  gru_fused_kernel<<<grid, block, 0, stream>>>(rnn_input, seq_len, kernel,
                                               rkernel, bias, out);
}
